// GATLayer_23725399343385
// MI455X (gfx1250) — matrix operand order repaired and checked by reading, not compiled, not run on hardware
//
#include <hip/hip_runtime.h>
#include <hip/hip_bf16.h>
#include <stdint.h>

#define B_   2
#define N_   2048
#define DIN_ 512
#define H_   8
#define F_   64
#define HF_  512
#define NEG_SLOPE_ 0.2f

typedef __attribute__((ext_vector_type(16))) __bf16 v16bf;
typedef __attribute__((ext_vector_type(8)))  float  v8f;

union FragB16 {
  v16bf v;
  uint4 u[2];
  unsigned short s[16];
};

__device__ __forceinline__ unsigned short f2bf(float f) {
  unsigned int u = __float_as_uint(f);
  u += 0x7FFFu + ((u >> 16) & 1u);          // round-to-nearest-even
  return (unsigned short)(u >> 16);
}
__device__ __forceinline__ float bf2f(unsigned short s) {
  return __uint_as_float(((unsigned int)s) << 16);
}
__device__ __forceinline__ float lrelu(float x) {
  return x > 0.f ? x : NEG_SLOPE_ * x;
}

// ---------------- pack kernels ----------------
__global__ void pack_nodes_kernel(const float* __restrict__ nodes,
                                  unsigned short* __restrict__ nodesbf) {
  int t = blockIdx.x * 256 + threadIdx.x;              // B*N*DIN = 2097152
  nodesbf[t] = f2bf(nodes[t]);
}

__global__ void transpose_w_kernel(const float* __restrict__ W,
                                   unsigned short* __restrict__ Wt) {
  int t = blockIdx.x * 256 + threadIdx.x;              // 512*512
  int n = t >> 9, k = t & 511;
  Wt[t] = f2bf(W[k * HF_ + n]);                        // Wt[n][k] = W[k][n]
}

// ---------------- GEMM: mapped = nodes @ W, stored transposed bf16 ----------------
// grid (256, 32) blocks of 32 threads. One wave -> one 16x16 tile of mapped.
__global__ void gemm_mapped_kernel(const unsigned short* __restrict__ nodesbf,
                                   const unsigned short* __restrict__ Wt,
                                   unsigned short* __restrict__ mT) {
  const int m0 = blockIdx.x * 16;                      // row in [0, B*N)
  const int n0 = blockIdx.y * 16;                      // col in [0, 512)
  const int lane = threadIdx.x;
  const int half = lane >> 4;
  const int l15  = lane & 15;

  v8f acc = {};
  const unsigned short* arow = nodesbf + (size_t)(m0 + l15) * DIN_;
  const unsigned short* bcol = Wt + (size_t)(n0 + l15) * DIN_;

  for (int k0 = 0; k0 < DIN_; k0 += 32) {
    FragB16 fa, fb;
    const unsigned short* ap = arow + k0 + half * 8;   // K: 0..7 / 8..15
    fa.u[0] = *(const uint4*)(ap);
    fa.u[1] = *(const uint4*)(ap + 16);                // K: 16..23 / 24..31
    const unsigned short* bp = bcol + k0 + half * 8;   // K: 0..7 / 8..15
    fb.u[0] = *(const uint4*)(bp);
    fb.u[1] = *(const uint4*)(bp + 16);
    acc = __builtin_amdgcn_wmma_f32_16x16x32_bf16(false, fa.v, false, fb.v,
                                                  (short)0, acc, false, false);
  }

  // epilogue: mapped(b, j, h, f) -> mappedT[b][h][f][j] in bf16
  const int b  = m0 >> 11;
  const int j0 = m0 & (N_ - 1);
  const int h  = n0 >> 6;
  const int f  = (n0 & 63) + l15;
  unsigned int p0 = (unsigned)f2bf(acc[0]) | ((unsigned)f2bf(acc[1]) << 16);
  unsigned int p1 = (unsigned)f2bf(acc[2]) | ((unsigned)f2bf(acc[3]) << 16);
  unsigned int p2 = (unsigned)f2bf(acc[4]) | ((unsigned)f2bf(acc[5]) << 16);
  unsigned int p3 = (unsigned)f2bf(acc[6]) | ((unsigned)f2bf(acc[7]) << 16);
  uint4 st; st.x = p0; st.y = p1; st.z = p2; st.w = p3;
  unsigned short* dst =
      mT + ((size_t)(b * H_ + h) * F_ + f) * N_ + j0 + half * 8;   // 8 consecutive j
  *(uint4*)dst = st;
}

// ---------------- el/er = mapped . a_l / a_r ----------------
__global__ void elr_kernel(const unsigned short* __restrict__ mT,
                           const float* __restrict__ a,
                           float* __restrict__ el, float* __restrict__ er) {
  int t = blockIdx.x * 256 + threadIdx.x;              // B*H*N = 32768
  int j = t & (N_ - 1);
  int bh = t >> 11;                                    // b*8+h
  const unsigned short* mp = mT + (size_t)bh * F_ * N_ + j;
  float sl = 0.f, sr = 0.f;
  #pragma unroll 8
  for (int f = 0; f < F_; ++f) {
    float v = bf2f(mp[(size_t)f * N_]);
    sl += v * a[f];
    sr += v * a[F_ + f];
  }
  el[t] = sl;
  er[t] = sr;
}

// ---------------- per-row softmax stats: m (row max score), rs = 1/sum ----------------
__global__ void stats_kernel(const int* __restrict__ edges,
                             const float* __restrict__ el,
                             const float* __restrict__ er,
                             float* __restrict__ m_out,
                             float* __restrict__ rs_out) {
  __shared__ float red[H_ * 256];
  __shared__ float Ms[H_], Es[H_];
  const int bi = blockIdx.x;                           // b*N + i
  const int b = bi >> 11, i = bi & (N_ - 1);
  const int tid = threadIdx.x;
  const int* erow = edges + (size_t)bi * N_;

  float mx[H_];
  #pragma unroll
  for (int hh = 0; hh < H_; ++hh) mx[hh] = -1e30f;
  for (int j = tid; j < N_; j += 256) {
    if (erow[j]) {
      #pragma unroll
      for (int hh = 0; hh < H_; ++hh)
        mx[hh] = fmaxf(mx[hh], er[(b * H_ + hh) * N_ + j]);
    }
  }
  #pragma unroll
  for (int hh = 0; hh < H_; ++hh) red[hh * 256 + tid] = mx[hh];
  __syncthreads();
  for (int s = 128; s > 0; s >>= 1) {
    if (tid < s) {
      #pragma unroll
      for (int hh = 0; hh < H_; ++hh)
        red[hh * 256 + tid] = fmaxf(red[hh * 256 + tid], red[hh * 256 + tid + s]);
    }
    __syncthreads();
  }
  if (tid < H_) {
    float elv = el[(b * H_ + tid) * N_ + i];
    Es[tid] = elv;
    Ms[tid] = lrelu(elv + red[tid * 256]);             // lrelu monotone => max commutes
  }
  __syncthreads();

  float sm[H_];
  #pragma unroll
  for (int hh = 0; hh < H_; ++hh) sm[hh] = 0.f;
  for (int j = tid; j < N_; j += 256) {
    if (erow[j]) {
      #pragma unroll
      for (int hh = 0; hh < H_; ++hh) {
        float sc = lrelu(Es[hh] + er[(b * H_ + hh) * N_ + j]);
        sm[hh] += __expf(sc - Ms[hh]);
      }
    }
  }
  #pragma unroll
  for (int hh = 0; hh < H_; ++hh) red[hh * 256 + tid] = sm[hh];
  __syncthreads();
  for (int s = 128; s > 0; s >>= 1) {
    if (tid < s) {
      #pragma unroll
      for (int hh = 0; hh < H_; ++hh)
        red[hh * 256 + tid] += red[hh * 256 + tid + s];
    }
    __syncthreads();
  }
  if (tid < H_) {
    m_out[(b * H_ + tid) * N_ + i]  = Ms[tid];
    rs_out[(b * H_ + tid) * N_ + i] = 1.0f / red[tid * 256];
  }
}

// ---------------- fused aggregation: out = sigmoid(mean_h (attn @ mapped)) ----------------
// grid = B*(N/32) workgroups, 256 threads = 8 waves, wave w = head w.
// Each wave handles TWO 16-row i-tiles (32 rows) so every B-fragment load
// feeds 2 WMMAs: 8 wmma per 8x b128 B-loads per 32-j chunk.
__global__ void agg_kernel(const int* __restrict__ edges,
                           const float* __restrict__ el,
                           const float* __restrict__ er,
                           const float* __restrict__ mrow,
                           const float* __restrict__ rsrow,
                           const unsigned short* __restrict__ mT,
                           float* __restrict__ out) {
  __shared__ float redv[H_][32 * F_];                  // 64 KB
  const int wg = blockIdx.x;                           // B*(N/32) = 128
  const int b  = wg >> 6;
  const int i0 = (wg & 63) << 5;                       // 32-row tile base
  const int tid  = threadIdx.x;
  const int h    = tid >> 5;
  const int lane = tid & 31;
  const int half = lane >> 4;
  const int l15  = lane & 15;

  const int bh = b * H_ + h;
  // tile 0: rows i0 + l15 ; tile 1: rows i0 + 16 + l15
  const float el0 = el[bh * N_ + i0 + l15];
  const float m0  = mrow[bh * N_ + i0 + l15];
  const float rs0 = rsrow[bh * N_ + i0 + l15];
  const float el1 = el[bh * N_ + i0 + 16 + l15];
  const float m1  = mrow[bh * N_ + i0 + 16 + l15];
  const float rs1 = rsrow[bh * N_ + i0 + 16 + l15];
  const float* erh = er + (size_t)bh * N_;
  const int* eb0 = edges + ((size_t)b * N_ + i0 + l15) * N_;
  const int* eb1 = eb0 + (size_t)16 * N_;
  const unsigned short* mTh = mT + (size_t)bh * F_ * N_;

  v8f a0c0 = {}, a0c1 = {}, a0c2 = {}, a0c3 = {};
  v8f a1c0 = {}, a1c1 = {}, a1c2 = {}, a1c3 = {};

  for (int j0 = 0; j0 < N_; j0 += 32) {
    const int ka = j0 + half * 8;                      // K 0..7 / 8..15
    const int kb = ka + 16;                            // K 16..23 / 24..31
    float4 e0 = *(const float4*)(erh + ka);
    float4 e1 = *(const float4*)(erh + ka + 4);
    float4 e2 = *(const float4*)(erh + kb);
    float4 e3 = *(const float4*)(erh + kb + 4);
    float ee[16] = {e0.x, e0.y, e0.z, e0.w, e1.x, e1.y, e1.z, e1.w,
                    e2.x, e2.y, e2.z, e2.w, e3.x, e3.y, e3.z, e3.w};

    int4 g0 = *(const int4*)(eb0 + ka);
    int4 g1 = *(const int4*)(eb0 + ka + 4);
    int4 g2 = *(const int4*)(eb0 + kb);
    int4 g3 = *(const int4*)(eb0 + kb + 4);
    int gg0[16] = {g0.x, g0.y, g0.z, g0.w, g1.x, g1.y, g1.z, g1.w,
                   g2.x, g2.y, g2.z, g2.w, g3.x, g3.y, g3.z, g3.w};
    int4 q0 = *(const int4*)(eb1 + ka);
    int4 q1 = *(const int4*)(eb1 + ka + 4);
    int4 q2 = *(const int4*)(eb1 + kb);
    int4 q3 = *(const int4*)(eb1 + kb + 4);
    int gg1[16] = {q0.x, q0.y, q0.z, q0.w, q1.x, q1.y, q1.z, q1.w,
                   q2.x, q2.y, q2.z, q2.w, q3.x, q3.y, q3.z, q3.w};

    FragB16 fa0, fa1;
    #pragma unroll
    for (int k = 0; k < 16; ++k) {
      float w0 = 0.f, w1 = 0.f;
      if (gg0[k]) w0 = __expf(lrelu(el0 + ee[k]) - m0) * rs0;
      if (gg1[k]) w1 = __expf(lrelu(el1 + ee[k]) - m1) * rs1;
      fa0.s[k] = f2bf(w0);
      fa1.s[k] = f2bf(w1);
    }

    const unsigned short* bp0 = mTh + (size_t)l15 * N_ + j0 + half * 8;
    FragB16 fb;
    fb.u[0] = *(const uint4*)(bp0);
    fb.u[1] = *(const uint4*)(bp0 + 16);
    a0c0 = __builtin_amdgcn_wmma_f32_16x16x32_bf16(false, fa0.v, false, fb.v,
                                                   (short)0, a0c0, false, false);
    a1c0 = __builtin_amdgcn_wmma_f32_16x16x32_bf16(false, fa1.v, false, fb.v,
                                                   (short)0, a1c0, false, false);
    const unsigned short* bp1 = bp0 + (size_t)16 * N_;
    fb.u[0] = *(const uint4*)(bp1);
    fb.u[1] = *(const uint4*)(bp1 + 16);
    a0c1 = __builtin_amdgcn_wmma_f32_16x16x32_bf16(false, fa0.v, false, fb.v,
                                                   (short)0, a0c1, false, false);
    a1c1 = __builtin_amdgcn_wmma_f32_16x16x32_bf16(false, fa1.v, false, fb.v,
                                                   (short)0, a1c1, false, false);
    const unsigned short* bp2 = bp0 + (size_t)32 * N_;
    fb.u[0] = *(const uint4*)(bp2);
    fb.u[1] = *(const uint4*)(bp2 + 16);
    a0c2 = __builtin_amdgcn_wmma_f32_16x16x32_bf16(false, fa0.v, false, fb.v,
                                                   (short)0, a0c2, false, false);
    a1c2 = __builtin_amdgcn_wmma_f32_16x16x32_bf16(false, fa1.v, false, fb.v,
                                                   (short)0, a1c2, false, false);
    const unsigned short* bp3 = bp0 + (size_t)48 * N_;
    fb.u[0] = *(const uint4*)(bp3);
    fb.u[1] = *(const uint4*)(bp3 + 16);
    a0c3 = __builtin_amdgcn_wmma_f32_16x16x32_bf16(false, fa0.v, false, fb.v,
                                                   (short)0, a0c3, false, false);
    a1c3 = __builtin_amdgcn_wmma_f32_16x16x32_bf16(false, fa1.v, false, fb.v,
                                                   (short)0, a1c3, false, false);
  }

  // per-head tiles -> LDS (fixed-order cross-head reduction => deterministic)
  #pragma unroll
  for (int r = 0; r < 8; ++r) {
    int row = r + half * 8;
    redv[h][row * F_ +  0 + l15] = a0c0[r];
    redv[h][row * F_ + 16 + l15] = a0c1[r];
    redv[h][row * F_ + 32 + l15] = a0c2[r];
    redv[h][row * F_ + 48 + l15] = a0c3[r];
    redv[h][(row + 16) * F_ +  0 + l15] = a1c0[r];
    redv[h][(row + 16) * F_ + 16 + l15] = a1c1[r];
    redv[h][(row + 16) * F_ + 32 + l15] = a1c2[r];
    redv[h][(row + 16) * F_ + 48 + l15] = a1c3[r];
  }
  __syncthreads();

  for (int idx = tid; idx < 32 * F_; idx += 256) {
    float s = 0.f;
    #pragma unroll
    for (int w = 0; w < H_; ++w) s += redv[w][idx];
    s *= (1.0f / H_);
    int row = idx >> 6, f = idx & 63;
    out[((size_t)b * N_ + i0 + row) * F_ + f] = 1.0f / (1.0f + __expf(-s));
  }
}

// ---------------- launch ----------------
extern "C" void kernel_launch(void* const* d_in, const int* in_sizes, int n_in,
                              void* d_out, int out_size, void* d_ws, size_t ws_size,
                              hipStream_t stream) {
  const float* nodes = (const float*)d_in[0];
  const int*   edges = (const int*)d_in[1];
  const float* W     = (const float*)d_in[2];
  const float* a     = (const float*)d_in[3];
  float*       out   = (float*)d_out;

  char* ws = (char*)d_ws;
  unsigned short* nodesbf = (unsigned short*)ws;                             // 4.0 MB
  unsigned short* Wt      = (unsigned short*)(ws + (size_t)4194304);         // 0.5 MB
  unsigned short* mT      = (unsigned short*)(ws + (size_t)4718592);         // 4.0 MB
  float* el  = (float*)(ws + (size_t)8912896);                              // 128 KB
  float* er  = el + B_ * H_ * N_;
  float* mr  = er + B_ * H_ * N_;
  float* rsr = mr + B_ * H_ * N_;

  pack_nodes_kernel<<<(B_ * N_ * DIN_) / 256, 256, 0, stream>>>(nodes, nodesbf);
  transpose_w_kernel<<<(DIN_ * HF_) / 256, 256, 0, stream>>>(W, Wt);
  gemm_mapped_kernel<<<dim3((B_ * N_) / 16, HF_ / 16), 32, 0, stream>>>(nodesbf, Wt, mT);
  elr_kernel<<<(B_ * H_ * N_) / 256, 256, 0, stream>>>(mT, a, el, er);
  stats_kernel<<<B_ * N_, 256, 0, stream>>>(edges, el, er, mr, rsr);
  agg_kernel<<<B_ * (N_ / 32), 256, 0, stream>>>(edges, el, er, mr, rsr, mT, out);
}
